// GAT_79035988181208
// MI455X (gfx1250) — matrix operand order repaired and checked by reading, not compiled, not run on hardware
//
#include <hip/hip_runtime.h>
#include <hip/hip_bf16.h>
#include <math.h>

#define NN     50000
#define EE     200000
#define ETOT   250000    // EE + NN self-loops
#define HID    64
#define HEADS  8
#define F1     512       // HEADS*HID
#define VOCAB  100
#define VP     112       // vocab padded to multiple of 16 (GEMM1 M, acc stride)
#define NEG    0.2f
#define EPSV   1e-16f

typedef __attribute__((ext_vector_type(16))) _Float16 v16h;
typedef __attribute__((ext_vector_type(8)))  float    v8f;

// ---------- utility: monotonic-int float atomic max (mem init'd to -inf) ----
__device__ __forceinline__ void atomicMaxFloat(float* a, float v) {
    if (v >= 0.f) atomicMax((int*)a, __float_as_int(v));
    else          atomicMin((unsigned int*)a, __float_as_uint(v));
}

// ---------- fill --------------------------------------------------------------
__global__ __launch_bounds__(256) void fill_f32(float* __restrict__ p, size_t n, float v) {
    size_t i = (size_t)blockIdx.x * 256 + threadIdx.x;
    if (i < n) p[i] = v;
}

// ---------- fp32 -> fp16 staging (emb padded to VP rows, Wl1/Wr1) -------------
__global__ __launch_bounds__(256) void prep_half(const float* __restrict__ emb,
                                                 const float* __restrict__ Wl1,
                                                 const float* __restrict__ Wr1,
                                                 _Float16* __restrict__ embH,
                                                 _Float16* __restrict__ WlH,
                                                 _Float16* __restrict__ WrH) {
    int i = blockIdx.x * 256 + threadIdx.x;
    if (i < HID * F1) { WlH[i] = (_Float16)Wl1[i]; WrH[i] = (_Float16)Wr1[i]; }
    if (i < VP * HID) {
        int r = i >> 6;
        embH[i] = (r < VOCAB) ? (_Float16)emb[i] : (_Float16)0.f;
    }
}

// ---------- GEMM1: hl = embPad @ Wl1 + bl1 ; hr = embPad @ Wr1 + br1 ----------
// [VP,64] @ [64,512]; one 16x16 output tile per wave; 448 tile-waves total.
__global__ __launch_bounds__(256) void gemm1_wmma(const _Float16* __restrict__ embH,
                                                  const _Float16* __restrict__ WlH,
                                                  const _Float16* __restrict__ WrH,
                                                  const float* __restrict__ bl1,
                                                  const float* __restrict__ br1,
                                                  float* __restrict__ hl,
                                                  float* __restrict__ hr,
                                                  _Float16* __restrict__ hlH) {
    int wid  = blockIdx.x * 8 + (threadIdx.x >> 5);      // 0..447 (grid exact)
    int lane = threadIdx.x & 31;
    int matSel = (wid >= 224);
    int t  = wid - matSel * 224;
    int tm = t >> 5;                                      // 0..6
    int tn = t & 31;                                      // 0..31
    const _Float16* W    = matSel ? WrH : WlH;
    const float*    bias = matSel ? br1 : bl1;
    float*          H    = matSel ? hr  : hl;

    int m  = lane & 15;
    int hi = lane >> 4;
    int arow = tm * 16 + m;           // A row (vocab)
    int bcol = tn * 16 + m;           // B/D column

    v8f c;
    float bv = bias[bcol];
    #pragma unroll
    for (int i = 0; i < 8; ++i) c[i] = bv;

    #pragma unroll
    for (int kb = 0; kb < HID; kb += 32) {
        v16h a, b;
        #pragma unroll
        for (int j = 0; j < 16; ++j) {
            int ka = kb + j + hi * 16;                          // A: contiguous k = 16h + s
            a[j] = embH[arow * HID + ka];
            int kk = kb + j + hi * 16;                          // B: 32x16 layout
            b[j] = W[kk * F1 + bcol];
        }
        c = __builtin_amdgcn_wmma_f32_16x16x32_f16(false, a, false, b, (short)0, c, false, false);
    }

    #pragma unroll
    for (int r = 0; r < 8; ++r) {
        int rr = tm * 16 + r + hi * 8;                          // D: M = r + hi*8
        float v = c[r];
        H[rr * F1 + bcol] = v;
        if (!matSel) hlH[rr * F1 + bcol] = (_Float16)v;         // fp16 copy for GEMM2 B
    }
}

// ---------- edge pass 1: attention logits + per-dst running max ---------------
// One wave per edge; 4 lanes per head, 16 channels per lane.
__global__ __launch_bounds__(256) void edge_logits(const int* __restrict__ x,
                                                   const int* __restrict__ ei,
                                                   const float* __restrict__ hl,
                                                   const float* __restrict__ hr,
                                                   const float* __restrict__ att1,
                                                   float* __restrict__ e1,
                                                   float* __restrict__ m1) {
    int wid = blockIdx.x * 8 + (threadIdx.x >> 5);
    if (wid >= ETOT) return;
    int lane = threadIdx.x & 31;
    int h = lane >> 2, sub = lane & 3;
    int src, dst;
    if (wid < EE) { src = ei[wid]; dst = ei[EE + wid]; } else { src = dst = wid - EE; }
    int xs = x[src], xd = x[dst];
    const float4* A  = (const float4*)(hl   + xs * F1 + h * HID + sub * 16);
    const float4* B  = (const float4*)(hr   + xd * F1 + h * HID + sub * 16);
    const float4* At = (const float4*)(att1 +           h * HID + sub * 16);
    float s = 0.f;
    #pragma unroll
    for (int i = 0; i < 4; ++i) {
        float4 a = A[i], b = B[i], w = At[i];
        float v;
        v = a.x + b.x; v = v > 0.f ? v : NEG * v; s += v * w.x;
        v = a.y + b.y; v = v > 0.f ? v : NEG * v; s += v * w.y;
        v = a.z + b.z; v = v > 0.f ? v : NEG * v; s += v * w.z;
        v = a.w + b.w; v = v > 0.f ? v : NEG * v; s += v * w.w;
    }
    s += __shfl_xor(s, 1, 4);
    s += __shfl_xor(s, 2, 4);
    if (sub == 0) {
        e1[wid * HEADS + h] = s;
        atomicMaxFloat(&m1[dst * HEADS + h], s);
    }
}

// ---------- edge pass 2: vocab-factorized aggregation weights ------------------
// acc[(dst*8+h)*VP + x[src]] += exp(e - max); den1[dst,h] += same.
__global__ __launch_bounds__(256) void edge_accum(const int* __restrict__ x,
                                                  const int* __restrict__ ei,
                                                  const float* __restrict__ e1,
                                                  const float* __restrict__ m1,
                                                  float* __restrict__ den1,
                                                  float* __restrict__ acc) {
    int e = blockIdx.x * 256 + threadIdx.x;
    if (e >= ETOT) return;
    int src, dst;
    if (e < EE) { src = ei[e]; dst = ei[EE + e]; } else { src = dst = e - EE; }
    int xs = x[src];
    #pragma unroll
    for (int h = 0; h < HEADS; ++h) {
        float p = __expf(e1[e * HEADS + h] - m1[dst * HEADS + h]);
        atomicAdd(&den1[dst * HEADS + h], p);
        atomicAdd(&acc[(size_t)(dst * HEADS + h) * VP + xs], p);
    }
}

// ---------- GEMM2: h1 = ELU( (acc @ hl_head) / den + bias1 ) per head ----------
// Block = 16-node tile, wave w = head w. A: [16,112→128] f32→f16, B: hlH cols.
__global__ __launch_bounds__(256) void gemm2_wmma(const float* __restrict__ acc,
                                                  const _Float16* __restrict__ hlH,
                                                  const float* __restrict__ den1,
                                                  const float* __restrict__ bias1,
                                                  float* __restrict__ h1) {
    int tile = blockIdx.x;                  // 0..3124 (exact: 3125*16 == 50000)
    int h    = threadIdx.x >> 5;            // head
    int lane = threadIdx.x & 31;
    int m  = lane & 15;
    int hi = lane >> 4;
    int node = tile * 16 + m;
    const float* arow = acc + (size_t)(node * HEADS + h) * VP;

    // preload the 4 A fragments (shared across all 4 column tiles)
    v16h a4[4];
    #pragma unroll
    for (int ks = 0; ks < 4; ++ks) {
        int kb = ks * 32;
        #pragma unroll
        for (int j = 0; j < 16; ++j) {
            int ka = kb + j + hi * 16;
            a4[ks][j] = (ka < VP) ? (_Float16)arow[ka] : (_Float16)0.f;
        }
    }

    #pragma unroll
    for (int ct = 0; ct < 4; ++ct) {
        int coln = h * HID + ct * 16 + m;   // global channel (B col, D col)
        v8f c = {};
        #pragma unroll
        for (int ks = 0; ks < 4; ++ks) {
            int kb = ks * 32;
            v16h b;
            #pragma unroll
            for (int j = 0; j < 16; ++j) {
                int kk = kb + j + hi * 16;  // < 128, hlH padded/zeroed to 128 rows
                b[j] = hlH[kk * F1 + coln];
            }
            c = __builtin_amdgcn_wmma_f32_16x16x32_f16(false, a4[ks], false, b, (short)0, c, false, false);
        }
        #pragma unroll
        for (int r = 0; r < 8; ++r) {
            int nd = tile * 16 + r + hi * 8;
            float den = den1[nd * HEADS + h] + EPSV;
            float v = c[r] / den + bias1[coln];
            v = v > 0.f ? v : (__expf(v) - 1.f);          // ELU
            h1[(size_t)nd * F1 + coln] = v;
        }
    }
}

// ---------- layer-2 node transforms: xl2 = h1@Wl2+bl2, xr2 = h1@Wr2+br2 --------
__global__ __launch_bounds__(256) void node_matvec(const float* __restrict__ h1,
                                                   const float* __restrict__ Wl2,
                                                   const float* __restrict__ bl2,
                                                   const float* __restrict__ Wr2,
                                                   const float* __restrict__ br2,
                                                   float* __restrict__ xl2,
                                                   float* __restrict__ xr2) {
    int n = blockIdx.x * 8 + (threadIdx.x >> 5);
    if (n >= NN) return;
    int lane = threadIdx.x & 31;
    float l0 = 0.f, l1 = 0.f, r0 = 0.f, r1 = 0.f;
    #pragma unroll
    for (int i = 0; i < 16; ++i) {
        int c = lane + (i << 5);
        float v = h1[(size_t)n * F1 + c];
        float2 wl = ((const float2*)Wl2)[c];
        float2 wr = ((const float2*)Wr2)[c];
        l0 += v * wl.x; l1 += v * wl.y;
        r0 += v * wr.x; r1 += v * wr.y;
    }
    #pragma unroll
    for (int o = 16; o >= 1; o >>= 1) {
        l0 += __shfl_xor(l0, o, 32);
        l1 += __shfl_xor(l1, o, 32);
        r0 += __shfl_xor(r0, o, 32);
        r1 += __shfl_xor(r1, o, 32);
    }
    if (lane == 0) {
        xl2[n * 2 + 0] = l0 + bl2[0];
        xl2[n * 2 + 1] = l1 + bl2[1];
        xr2[n * 2 + 0] = r0 + br2[0];
        xr2[n * 2 + 1] = r1 + br2[1];
    }
}

// ---------- layer-2 edge logits + max ------------------------------------------
__global__ __launch_bounds__(256) void edge2_logits(const int* __restrict__ ei,
                                                    const float* __restrict__ xl2,
                                                    const float* __restrict__ xr2,
                                                    const float* __restrict__ att2,
                                                    float* __restrict__ e2,
                                                    float* __restrict__ m2) {
    int e = blockIdx.x * 256 + threadIdx.x;
    if (e >= ETOT) return;
    int src, dst;
    if (e < EE) { src = ei[e]; dst = ei[EE + e]; } else { src = dst = e - EE; }
    float2 a = ((const float2*)xl2)[src];
    float2 b = ((const float2*)xr2)[dst];
    float v0 = a.x + b.x; v0 = v0 > 0.f ? v0 : NEG * v0;
    float v1 = a.y + b.y; v1 = v1 > 0.f ? v1 : NEG * v1;
    float s = v0 * att2[0] + v1 * att2[1];
    e2[e] = s;
    atomicMaxFloat(&m2[dst], s);
}

// ---------- layer-2 edge aggregation -------------------------------------------
__global__ __launch_bounds__(256) void edge2_accum(const int* __restrict__ ei,
                                                   const float* __restrict__ e2,
                                                   const float* __restrict__ m2,
                                                   const float* __restrict__ xl2,
                                                   float* __restrict__ den2,
                                                   float* __restrict__ agg2) {
    int e = blockIdx.x * 256 + threadIdx.x;
    if (e >= ETOT) return;
    int src, dst;
    if (e < EE) { src = ei[e]; dst = ei[EE + e]; } else { src = dst = e - EE; }
    float p = __expf(e2[e] - m2[dst]);
    atomicAdd(&den2[dst], p);
    float2 a = ((const float2*)xl2)[src];
    atomicAdd(&agg2[dst * 2 + 0], p * a.x);
    atomicAdd(&agg2[dst * 2 + 1], p * a.y);
}

// ---------- final normalize + bias ---------------------------------------------
__global__ __launch_bounds__(256) void finalize(const float* __restrict__ agg2,
                                                const float* __restrict__ den2,
                                                const float* __restrict__ bias2,
                                                float* __restrict__ out) {
    int i = blockIdx.x * 256 + threadIdx.x;
    if (i >= 2 * NN) return;
    int n = i >> 1, j = i & 1;
    out[i] = agg2[i] / (den2[n] + EPSV) + bias2[j];
}

// ================================================================================
extern "C" void kernel_launch(void* const* d_in, const int* in_sizes, int n_in,
                              void* d_out, int out_size, void* d_ws, size_t ws_size,
                              hipStream_t stream) {
    (void)in_sizes; (void)n_in; (void)out_size; (void)ws_size;
    const int*   x     = (const int*)  d_in[0];
    const int*   ei    = (const int*)  d_in[1];
    const float* emb   = (const float*)d_in[2];
    const float* Wl1   = (const float*)d_in[3];
    const float* bl1   = (const float*)d_in[4];
    const float* Wr1   = (const float*)d_in[5];
    const float* br1   = (const float*)d_in[6];
    const float* att1  = (const float*)d_in[7];
    const float* bias1 = (const float*)d_in[8];
    const float* Wl2   = (const float*)d_in[9];
    const float* bl2   = (const float*)d_in[10];
    const float* Wr2   = (const float*)d_in[11];
    const float* br2   = (const float*)d_in[12];
    const float* att2  = (const float*)d_in[13];
    const float* bias2 = (const float*)d_in[14];
    float* out = (float*)d_out;

    // ---- workspace carve (all offsets 256B-aligned) ----
    char* base = (char*)d_ws;
    size_t off = 0;
    auto carve = [&](size_t bytes) -> char* {
        char* p = base + off;
        off += (bytes + 255) & ~(size_t)255;
        return p;
    };
    // zero-init region (contiguous)
    char* zstart = base + off;
    float*    den1 = (float*)   carve((size_t)NN * HEADS * 4);
    float*    acc  = (float*)   carve((size_t)NN * HEADS * VP * 4);
    float*    den2 = (float*)   carve((size_t)NN * 4);
    float*    agg2 = (float*)   carve((size_t)NN * 2 * 4);
    _Float16* hlH  = (_Float16*)carve((size_t)128 * F1 * 2);   // K padded to 128
    size_t zeroFloats = (size_t)((base + off) - zstart) / 4;
    // -inf region (contiguous)
    char* nstart = base + off;
    float* m1 = (float*)carve((size_t)NN * HEADS * 4);
    float* m2 = (float*)carve((size_t)NN * 4);
    size_t negFloats = (size_t)((base + off) - nstart) / 4;
    // rest
    float*    e1   = (float*)   carve((size_t)ETOT * HEADS * 4);
    float*    e2   = (float*)   carve((size_t)ETOT * 4);
    float*    hl   = (float*)   carve((size_t)VP * F1 * 4);
    float*    hr   = (float*)   carve((size_t)VP * F1 * 4);
    float*    h1   = (float*)   carve((size_t)NN * F1 * 4);
    float*    xl2  = (float*)   carve((size_t)NN * 2 * 4);
    float*    xr2  = (float*)   carve((size_t)NN * 2 * 4);
    _Float16* embH = (_Float16*)carve((size_t)VP * HID * 2);
    _Float16* WlH  = (_Float16*)carve((size_t)HID * F1 * 2);
    _Float16* WrH  = (_Float16*)carve((size_t)HID * F1 * 2);

    // ---- pipeline (stream-ordered, graph-capture safe) ----
    fill_f32<<<(unsigned)((zeroFloats + 255) / 256), 256, 0, stream>>>((float*)zstart, zeroFloats, 0.f);
    fill_f32<<<(unsigned)((negFloats  + 255) / 256), 256, 0, stream>>>((float*)nstart, negFloats, -INFINITY);
    prep_half<<<(HID * F1 + 255) / 256, 256, 0, stream>>>(emb, Wl1, Wr1, embH, WlH, WrH);
    gemm1_wmma<<<56, 256, 0, stream>>>(embH, WlH, WrH, bl1, br1, hl, hr, hlH);
    edge_logits<<<ETOT / 8, 256, 0, stream>>>(x, ei, hl, hr, att1, e1, m1);
    edge_accum<<<(ETOT + 255) / 256, 256, 0, stream>>>(x, ei, e1, m1, den1, acc);
    gemm2_wmma<<<NN / 16, 256, 0, stream>>>(acc, hlH, den1, bias1, h1);
    node_matvec<<<NN / 8, 256, 0, stream>>>(h1, Wl2, bl2, Wr2, br2, xl2, xr2);
    edge2_logits<<<(ETOT + 255) / 256, 256, 0, stream>>>(ei, xl2, xr2, att2, e2, m2);
    edge2_accum<<<(ETOT + 255) / 256, 256, 0, stream>>>(ei, e2, m2, xl2, den2, agg2);
    finalize<<<(2 * NN + 255) / 256, 256, 0, stream>>>(agg2, den2, bias2, out);
}
